// MultiLatentAttention_33526514713085
// MI455X (gfx1250) — matrix operand order repaired and checked by reading, not compiled, not run on hardware
//
#include <hip/hip_runtime.h>
#include <stdint.h>

// ---------------------------------------------------------------------------
// MultiLatentAttention for MI455X (gfx1250), wave32 + WMMA bf16 +
// async global->LDS staging (ASYNCcnt) with double buffering.
//
// Shapes: B=4, T=4096, C=1024, H=16, D=64, L=64.
// Causal mask is t<=l with l<64 -> attention only reads the first 64 tokens
// of K/V per batch, so K/V projections shrink to 256 rows.
// All GEMM B operands are pre-transposed at bf16-conversion time so every
// LDS staging transfer is a 16-byte row-major chunk.
// ---------------------------------------------------------------------------

typedef __attribute__((ext_vector_type(16))) __bf16 bf16x16;
typedef __attribute__((ext_vector_type(8)))  float  f32x8;

#define B_  4
#define T_  4096
#define C_  1024
#define H_  16
#define D_  64
#define L_  64
#define ROWS_ (B_ * T_)          // 16384

#if defined(__AMDGCN__) && __has_builtin(__builtin_amdgcn_global_load_async_to_lds_b128)
#define HAVE_ASYNC 1
typedef int v4i_ __attribute__((vector_size(16)));
typedef __attribute__((address_space(1))) v4i_ as1_v4i;
typedef __attribute__((address_space(3))) v4i_ as3_v4i;
#else
#define HAVE_ASYNC 0
#endif

__device__ __forceinline__ void cp16_async(void* lds, const void* g) {
#if HAVE_ASYNC
  // GLOBAL_LOAD_ASYNC_TO_LDS_B128: per-lane 16B memory->LDS, ASYNCcnt tracked
  __builtin_amdgcn_global_load_async_to_lds_b128(
      (as1_v4i*)g, (as3_v4i*)lds, 0, 0);
#else
  *(uint4*)lds = *(const uint4*)g;
#endif
}

__device__ __forceinline__ void wait_async_all() {
#if HAVE_ASYNC
#if __has_builtin(__builtin_amdgcn_s_wait_asynccnt)
  __builtin_amdgcn_s_wait_asynccnt(0);
#else
  asm volatile("s_wait_asynccnt 0x0" ::: "memory");
#endif
#endif
}

__device__ __forceinline__ uint16_t f32_to_bf16_rne(float f) {
  union { float f; uint32_t u; } cv; cv.f = f;
  uint32_t u = cv.u;
  uint32_t r;
  if ((u & 0x7F800000u) == 0x7F800000u) r = u | 0x00400000u;   // quiet NaN/Inf
  else                                  r = u + 0x7FFFu + ((u >> 16) & 1u);
  return (uint16_t)(r >> 16);
}

// ------------------------- fp32 -> bf16 conversion -------------------------
__global__ void mla_cvt_bf16(const float* __restrict__ in,
                             uint16_t* __restrict__ out, int n) {
  int i = blockIdx.x * blockDim.x + threadIdx.x;
  int stride = gridDim.x * blockDim.x;
  for (; i < n; i += stride) out[i] = f32_to_bf16_rne(in[i]);
}

// fp32 [K x N] -> bf16 transposed [N x K] (weights; one-time cost)
__global__ void mla_cvt_bf16_T(const float* __restrict__ in,
                               uint16_t* __restrict__ out, int K, int N) {
  int i = blockIdx.x * blockDim.x + threadIdx.x;
  int stride = gridDim.x * blockDim.x;
  int total = K * N;
  for (; i < total; i += stride) {
    int k = i / N, n = i - k * N;
    out[(size_t)n * K + k] = f32_to_bf16_rne(in[i]);
  }
}

// ------------------- generic bf16 WMMA GEMM, fp32 output -------------------
// C[M x N] = A[M x K] * Bt[N x K]^T; A row-major bf16, Bt = B^T row-major.
// Block = 256 threads (8 waves). Block tile 128x128, wave tile 32x64.
// Double-buffered LDS, async global->LDS staging.
// kvRemap: logical row r -> physical A row (r/64)*4096 + (r%64).
#define BM 128
#define BN 128
#define BK 32

__global__ __launch_bounds__(256)
void mla_gemm_bf16(const uint16_t* __restrict__ A, int lda,
                   const uint16_t* __restrict__ Bt, int ldbt,
                   float* __restrict__ C, int ldc,
                   int M, int N, int K, int kvRemap) {
  __shared__ __align__(16) uint16_t As[2][BM][BK + 8];  // 80B rows, 16B align
  __shared__ __align__(16) uint16_t Bs[2][BN][BK + 8];  // Bs[n][k]

  const int tid  = threadIdx.x;
  const int lane = tid & 31;
  const int wid  = tid >> 5;       // 0..7
  const int wm   = wid >> 1;       // 0..3 : 32-row slabs
  const int wn   = wid & 1;        // 0..1 : 64-col slabs

  const int m0 = blockIdx.y * BM;
  const int n0 = blockIdx.x * BN;

  f32x8 acc[2][4];
#pragma unroll
  for (int i = 0; i < 2; ++i)
#pragma unroll
    for (int j = 0; j < 4; ++j)
#pragma unroll
      for (int e = 0; e < 8; ++e) acc[i][j][e] = 0.0f;

  const int mr   = lane & 15;      // fragment row/col within 16
  const int half = lane >> 4;

  // stage one K-step (A: 128x32, Bt: 128x32) into buffer `buf`
  auto kick = [&](int buf, int k0) {
#pragma unroll
    for (int c = tid; c < (BM * BK) / 8; c += 256) {
      int r = c >> 2, kk = (c & 3) * 8;
      int grow = m0 + r;
      int garow = kvRemap ? ((grow >> 6) * T_ + (grow & 63)) : grow;
      cp16_async(&As[buf][r][kk], A + (size_t)garow * lda + k0 + kk);
    }
#pragma unroll
    for (int c = tid; c < (BN * BK) / 8; c += 256) {
      int r = c >> 2, kk = (c & 3) * 8;
      cp16_async(&Bs[buf][r][kk], Bt + (size_t)(n0 + r) * ldbt + k0 + kk);
    }
  };

  const int nIter = K / BK;
  kick(0, 0);

  for (int it = 0; it < nIter; ++it) {
    const int cur = it & 1;
    wait_async_all();          // my async fills for tile `it` are done
    __syncthreads();           // everyone's fills are done
    if (it + 1 < nIter) kick(cur ^ 1, (it + 1) * BK);

    bf16x16 afrag[2], bfrag[4];
#pragma unroll
    for (int i = 0; i < 2; ++i) {
      int row = wm * 32 + i * 16 + mr;
      uint4* p = (uint4*)&afrag[i];
      p[0] = *(const uint4*)&As[cur][row][half * 8];      // K 0..7  / 8..15
      p[1] = *(const uint4*)&As[cur][row][16 + half * 8]; // K 16..23/24..31
    }
#pragma unroll
    for (int j = 0; j < 4; ++j) {
      int col = wn * 64 + j * 16 + mr;
      uint4* p = (uint4*)&bfrag[j];
      p[0] = *(const uint4*)&Bs[cur][col][half * 8];      // K 0..7  / 8..15
      p[1] = *(const uint4*)&Bs[cur][col][16 + half * 8]; // K 16..23/24..31
    }
#pragma unroll
    for (int i = 0; i < 2; ++i)
#pragma unroll
      for (int j = 0; j < 4; ++j)
        acc[i][j] = __builtin_amdgcn_wmma_f32_16x16x32_bf16(
            false, afrag[i], false, bfrag[j], (short)0, acc[i][j],
            false, false);
  }

  // store: VGPR r -> M=r (lanes 0-15), M=8+r (lanes 16-31); N = lane&15
  const int cn = lane & 15;
  const int ch = lane >> 4;
#pragma unroll
  for (int i = 0; i < 2; ++i)
#pragma unroll
    for (int j = 0; j < 4; ++j) {
      int gm = m0 + wm * 32 + i * 16 + ch * 8;
      int gn = n0 + wn * 64 + j * 16 + cn;
#pragma unroll
      for (int r = 0; r < 8; ++r)
        C[(size_t)(gm + r) * ldc + gn] = acc[i][j][r];
    }
}

// --------------------- attention core (fp32, 64x64x64) ---------------------
// One block per (b,h); 64 threads, thread l owns latent row l.
__global__ __launch_bounds__(64)
void mla_attn_core(const float* __restrict__ kproj,    // [256 x 1024]
                   const float* __restrict__ vproj,    // [256 x 1024]
                   const float* __restrict__ latent_q, // [1,L,H,D]
                   uint16_t* __restrict__ attn_out) {  // [B*H, L, D] bf16
  const int bh = blockIdx.x;
  const int b = bh >> 4, h = bh & 15;

  __shared__ float qs[64][64];
  __shared__ float ks[64][64];
  __shared__ float vs[64][64];

  const int l = threadIdx.x;          // also reused as token t for loading
  for (int d = 0; d < 64; ++d)
    qs[l][d] = latent_q[((size_t)l * H_ + h) * D_ + d];

  {  // k with interleaved RoPE (angle = t * theta^(-2i/D)); v plain
    const int t = l;
    const float* kr = kproj + (size_t)(b * 64 + t) * C_ + h * D_;
    const float* vr = vproj + (size_t)(b * 64 + t) * C_ + h * D_;
    const float ln_theta = 9.210340371976184f;   // ln(10000)
    for (int i = 0; i < 32; ++i) {
      float inv = __expf(-((float)(2 * i) / 64.0f) * ln_theta);
      float ang = (float)t * inv;
      float cc = __cosf(ang), ss = __sinf(ang);
      float k0 = kr[2 * i], k1 = kr[2 * i + 1];
      ks[t][2 * i]     = k0 * cc - k1 * ss;
      ks[t][2 * i + 1] = k0 * ss + k1 * cc;
      vs[t][2 * i]     = vr[2 * i];
      vs[t][2 * i + 1] = vr[2 * i + 1];
    }
  }
  __syncthreads();

  float p[64];
  float mx = -3.0e38f;
#pragma unroll
  for (int t = 0; t < 64; ++t) {
    float acc = 0.0f;
    for (int d = 0; d < 64; ++d) acc += qs[l][d] * ks[t][d];
    acc *= 0.125f;                     // 1/sqrt(64)
    p[t] = (t <= l) ? acc : -3.0e38f;
    mx = fmaxf(mx, p[t]);
  }
  float sum = 0.0f;
#pragma unroll
  for (int t = 0; t < 64; ++t) {
    float e = (t <= l) ? __expf(p[t] - mx) : 0.0f;
    p[t] = e;
    sum += e;
  }
  const float rinv = 1.0f / sum;

  for (int d = 0; d < 64; ++d) {
    float o = 0.0f;
#pragma unroll
    for (int t = 0; t < 64; ++t) o += p[t] * vs[t][d];
    attn_out[((size_t)bh * L_ + l) * D_ + d] = f32_to_bf16_rne(o * rinv);
  }
}

// --------------------- gate softmax (wave per token-head) -------------------
__global__ __launch_bounds__(256)
void mla_gate_softmax(const float* __restrict__ graw,   // [16384 x 1024]
                      uint16_t* __restrict__ gates,     // [16384 x 1024] bf16
                      int total) {                      // ROWS_*H_
  const int gid = blockIdx.x * 8 + (threadIdx.x >> 5);
  if (gid >= total) return;
  const int lane = threadIdx.x & 31;
  const int row = gid >> 4, h = gid & 15;

  const float* src = graw + (size_t)row * C_ + h * L_;
  float a = src[lane], b = src[lane + 32];
  float m = fmaxf(a, b);
#pragma unroll
  for (int off = 16; off; off >>= 1) m = fmaxf(m, __shfl_xor(m, off, 32));
  float ea = __expf(a - m), eb = __expf(b - m);
  float s = ea + eb;
#pragma unroll
  for (int off = 16; off; off >>= 1) s += __shfl_xor(s, off, 32);
  const float inv = 1.0f / s;

  uint16_t* dst = gates + (size_t)row * C_ + h * L_;
  dst[lane]      = f32_to_bf16_rne(ea * inv);
  dst[lane + 32] = f32_to_bf16_rne(eb * inv);
}

// ------------- gate-combine: weighted[t,d] = sum_l g[t,l]*ao[l,d] ----------
// Per (b,h): [4096 x 64] @ [64 x 64], WMMA bf16, bf16 output at column
// h*64+d of the [16384 x 1024] matrix the final GEMM consumes.
__global__ __launch_bounds__(256)
void mla_combine_wmma(const uint16_t* __restrict__ gates,   // [16384 x 1024]
                      const uint16_t* __restrict__ attn,    // [B*H, L, D]
                      uint16_t* __restrict__ wout) {        // [16384 x 1024]
  const int bh = blockIdx.y;
  const int b = bh >> 4, h = bh & 15;
  const int t0 = blockIdx.x * 128;

  __shared__ __align__(16) uint16_t As[128][72];   // full 128x64 gates slab
  __shared__ __align__(16) uint16_t Bs[64][72];    // Bs[d][l] = attn^T

  const int tid = threadIdx.x;
  const int lane = tid & 31;
  const int wid = tid >> 5;                        // 8 waves -> 16 rows each
  const int mr = lane & 15;
  const int half = lane >> 4;

  // async-stage the whole 128x64 A slab (16B chunks)
#pragma unroll
  for (int c = tid; c < (128 * 64) / 8; c += 256) {
    int r = c >> 3, kk = (c & 7) * 8;
    cp16_async(&As[r][kk],
               gates + (size_t)(b * T_ + t0 + r) * C_ + h * L_ + kk);
  }
  // Bt: Bs[d][l] = attn[bh][l][d] (tiny, sync transpose)
  for (int c = tid; c < 64 * 64; c += 256) {
    int d = c >> 6, l = c & 63;
    Bs[d][l] = attn[((size_t)bh * L_ + l) * D_ + d];
  }
  wait_async_all();
  __syncthreads();

  f32x8 acc[4];
#pragma unroll
  for (int j = 0; j < 4; ++j)
#pragma unroll
    for (int e = 0; e < 8; ++e) acc[j][e] = 0.0f;

#pragma unroll
  for (int k0 = 0; k0 < 64; k0 += 32) {
    bf16x16 af, bf[4];
    {
      int row = wid * 16 + mr;
      uint4* p = (uint4*)&af;
      p[0] = *(const uint4*)&As[row][k0 + half * 8];
      p[1] = *(const uint4*)&As[row][k0 + 16 + half * 8];
    }
#pragma unroll
    for (int j = 0; j < 4; ++j) {
      int col = j * 16 + mr;
      uint4* p = (uint4*)&bf[j];
      p[0] = *(const uint4*)&Bs[col][k0 + half * 8];
      p[1] = *(const uint4*)&Bs[col][k0 + 16 + half * 8];
    }
#pragma unroll
    for (int j = 0; j < 4; ++j)
      acc[j] = __builtin_amdgcn_wmma_f32_16x16x32_bf16(
          false, af, false, bf[j], (short)0, acc[j], false, false);
  }

  const int cn = lane & 15;
  const int ch = lane >> 4;
#pragma unroll
  for (int j = 0; j < 4; ++j) {
    int col = h * D_ + j * 16 + cn;
#pragma unroll
    for (int r = 0; r < 8; ++r) {
      int trow = t0 + wid * 16 + ch * 8 + r;
      wout[(size_t)(b * T_ + trow) * C_ + col] = f32_to_bf16_rne(acc[j][r]);
    }
  }
}

// ------------------------------- launcher ----------------------------------
extern "C" void kernel_launch(void* const* d_in, const int* in_sizes, int n_in,
                              void* d_out, int out_size, void* d_ws,
                              size_t ws_size, hipStream_t stream) {
  (void)in_sizes; (void)n_in; (void)out_size; (void)ws_size;
  const float* x  = (const float*)d_in[0];
  const float* lq = (const float*)d_in[1];
  const float* Wk = (const float*)d_in[2];
  const float* Wv = (const float*)d_in[3];
  const float* Wg = (const float*)d_in[4];
  const float* Wp = (const float*)d_in[5];
  float* out = (float*)d_out;

  char* ws = (char*)d_ws;
  size_t off = 0;
  auto take = [&](size_t bytes) -> void* {
    void* p = ws + off;
    off += (bytes + 255) & ~(size_t)255;
    return p;
  };
  uint16_t* xb    = (uint16_t*)take((size_t)ROWS_ * C_ * 2);  // 33.5 MB
  uint16_t* wkbT  = (uint16_t*)take((size_t)C_ * C_ * 2);     // transposed
  uint16_t* wvbT  = (uint16_t*)take((size_t)C_ * C_ * 2);
  uint16_t* wgbT  = (uint16_t*)take((size_t)C_ * C_ * 2);
  uint16_t* wpbT  = (uint16_t*)take((size_t)C_ * C_ * 2);
  float*    kproj = (float*)take((size_t)B_ * 64 * C_ * 4);   // 1 MB
  float*    vproj = (float*)take((size_t)B_ * 64 * C_ * 4);
  float*    graw  = (float*)take((size_t)ROWS_ * C_ * 4);     // 67 MB
  uint16_t* gatesb= (uint16_t*)take((size_t)ROWS_ * C_ * 2);  // 33.5 MB
  uint16_t* attnb = (uint16_t*)take((size_t)B_ * H_ * L_ * D_ * 2);
  uint16_t* wtdb  = xb;   // xb dead after gates/kv GEMMs; reuse region

  // 1) fp32 -> bf16 (x row-major; weights transposed for row-major staging)
  mla_cvt_bf16  <<<2048, 256, 0, stream>>>(x,  xb,  ROWS_ * C_);
  mla_cvt_bf16_T<<<512,  256, 0, stream>>>(Wk, wkbT, C_, C_);
  mla_cvt_bf16_T<<<512,  256, 0, stream>>>(Wv, wvbT, C_, C_);
  mla_cvt_bf16_T<<<512,  256, 0, stream>>>(Wg, wgbT, C_, C_);
  mla_cvt_bf16_T<<<512,  256, 0, stream>>>(Wp, wpbT, C_, C_);

  // 2) K/V projections: only first 64 tokens per batch (row remap), 256x1024
  {
    dim3 g(C_ / BN, (B_ * 64) / BM);
    mla_gemm_bf16<<<g, 256, 0, stream>>>(xb, C_, wkbT, C_, kproj, C_,
                                         B_ * 64, C_, C_, 1);
    mla_gemm_bf16<<<g, 256, 0, stream>>>(xb, C_, wvbT, C_, vproj, C_,
                                         B_ * 64, C_, C_, 1);
  }
  // 3) gate logits: 16384 x 1024
  {
    dim3 g(C_ / BN, ROWS_ / BM);
    mla_gemm_bf16<<<g, 256, 0, stream>>>(xb, C_, wgbT, C_, graw, C_,
                                         ROWS_, C_, C_, 0);
  }
  // 4) attention core (RoPE + masked softmax + AV), one block per (b,h)
  mla_attn_core<<<B_ * H_, 64, 0, stream>>>(kproj, vproj, lq, attnb);

  // 5) gate softmax over L per (token, head)
  mla_gate_softmax<<<(ROWS_ * H_) / 8, 256, 0, stream>>>(graw, gatesb,
                                                         ROWS_ * H_);
  // 6) gate-combine into [16384 x 1024] bf16 (reuses xb region)
  {
    dim3 g(T_ / 128, B_ * H_);
    mla_combine_wmma<<<g, 256, 0, stream>>>(gatesb, attnb, wtdb);
  }
  // 7) output projection: 16384 x 1024 fp32
  {
    dim3 g(C_ / BN, ROWS_ / BM);
    mla_gemm_bf16<<<g, 256, 0, stream>>>(wtdb, C_, wpbT, C_, out, C_,
                                         ROWS_, C_, C_, 0);
  }
}
